// CustomDecoder_58471684768447
// MI455X (gfx1250) — matrix operand order repaired and checked by reading, not compiled, not run on hardware
//
#include <hip/hip_runtime.h>

// ---------- types for WMMA ----------
typedef __attribute__((ext_vector_type(16))) __bf16 v16bf;
typedef __attribute__((ext_vector_type(8)))  float  v8f;
typedef __attribute__((ext_vector_type(4)))  int    v4i;

#define DEV static __device__ __forceinline__

constexpr int Bsz   = 512;
constexpr int SENC  = 1024;
constexpr int T     = 128;
constexpr int D     = 64;
constexpr int Lp    = 256;
constexpr int C     = 128;
constexpr int SLOTS = T + 1;   // 129

// ---------- CDNA5 async global->LDS path (guarded; sync fallback) ----------
#if defined(__HIP_DEVICE_COMPILE__) && __has_builtin(__builtin_amdgcn_global_load_async_to_lds_b128)
#define HAS_ASYNC 1
#else
#define HAS_ASYNC 0
#endif

DEV void async_cp16(const unsigned short* g, unsigned short* l) {
#if HAS_ASYNC
  __builtin_amdgcn_global_load_async_to_lds_b128(
      (__attribute__((address_space(1))) v4i*)g,
      (__attribute__((address_space(3))) v4i*)l, 0, 0);
#else
  *(uint4*)l = *(const uint4*)g;
#endif
}

template <int N> DEV void wait_async() {
#if HAS_ASYNC
 #if __has_builtin(__builtin_amdgcn_s_wait_asynccnt)
  __builtin_amdgcn_s_wait_asynccnt(N);
 #else
  asm volatile("s_wait_asynccnt %0" :: "i"(N) : "memory");
 #endif
#endif
}

// ---------- bf16 helpers (native converts -> v_cvt_*bf16*) ----------
DEV __bf16 f2bfh(float f) { return (__bf16)f; }
DEV unsigned short f2bf(float f) { return __builtin_bit_cast(unsigned short, (__bf16)f); }
DEV float lo16f(unsigned u) { union { unsigned u; float f; } v; v.u = u << 16;        return v.f; }
DEV float hi16f(unsigned u) { union { unsigned u; float f; } v; v.u = u & 0xFFFF0000u; return v.f; }

DEV float wred_max(float x) {
  #pragma unroll
  for (int o = 16; o; o >>= 1) x = fmaxf(x, __shfl_xor(x, o, 32));
  return x;
}
DEV float wred_sum(float x) {
  #pragma unroll
  for (int o = 16; o; o >>= 1) x += __shfl_xor(x, o, 32);
  return x;
}

// ============================================================================
// Kernel 1: one-off tiny precompute (pos K/V projections; slot-0 state).
// ============================================================================
__global__ void __launch_bounds__(64) k_init(
    const float* __restrict__ pos, const float* __restrict__ h0,
    const float* __restrict__ sa_w, const float* __restrict__ sa_b,
    float* __restrict__ posk, float* __restrict__ posv,
    float* __restrict__ H, float* __restrict__ Hk, float* __restrict__ Hv)
{
  __shared__ float row[D];
  int bid = blockIdx.x, n = threadIdx.x;
  if (bid < Lp) {
    row[n] = pos[bid * D + n];
    __syncthreads();
    float sk = 0.f, sv = 0.f;
    for (int k = 0; k < D; k++) {
      float r = row[k];
      sk += r * sa_w[(D     + n) * D + k];
      sv += r * sa_w[(2 * D + n) * D + k];
    }
    posk[bid * D + n] = sk;
    posv[bid * D + n] = sv;
  } else {
    int b = bid - Lp;
    row[n] = h0[n];
    __syncthreads();
    float sk = sa_b[D + n], sv = sa_b[2 * D + n];
    for (int k = 0; k < D; k++) {
      float r = row[k];
      sk += r * sa_w[(D     + n) * D + k];
      sv += r * sa_w[(2 * D + n) * D + k];
    }
    H [(size_t)b * SLOTS * D + n] = row[n];
    Hk[(size_t)b * SLOTS * D + n] = sk;
    Hv[(size_t)b * SLOTS * D + n] = sv;
  }
}

// ============================================================================
// Kernel 2: encoder prep (WMMA). layernorm(enc) -> Kimg/Vimg in bf16
// (134 MB total: resident in the 192 MB L2 for the 128 streaming passes).
// ============================================================================
__global__ void __launch_bounds__(32) k_encprep(
    const float* __restrict__ enc, const float* __restrict__ g, const float* __restrict__ beta,
    const float* __restrict__ ia_w, const float* __restrict__ ia_b,
    unsigned short* __restrict__ Kimg, unsigned short* __restrict__ Vimg)
{
  __shared__ float sg[D], sb[D];
  __shared__ unsigned short tile[16 * D];
  int lane = threadIdx.x;
  size_t row0 = (size_t)blockIdx.x * 16;

  sg[lane] = g[lane];  sg[lane + 32] = g[lane + 32];
  sb[lane] = beta[lane]; sb[lane + 32] = beta[lane + 32];
  __syncthreads();

  int r = lane & 15, hh = lane >> 4;
  const float* src = enc + (row0 + r) * D + hh * 32;
  float x[32], s = 0.f, sq = 0.f;
  #pragma unroll
  for (int i = 0; i < 32; i++) { x[i] = src[i]; s += x[i]; sq += x[i] * x[i]; }
  s  += __shfl_xor(s, 16, 32);
  sq += __shfl_xor(sq, 16, 32);
  float mean = s * (1.f / 64.f);
  float rstd = rsqrtf(sq * (1.f / 64.f) - mean * mean + 1e-5f);
  #pragma unroll
  for (int i = 0; i < 32; i++) {
    int e = hh * 32 + i;
    tile[r * D + e] = f2bf((x[i] - mean) * rstd * sg[e] + sb[e]);
  }
  __syncthreads();

  v16bf a[2];   // ISA 16-bit A 16x32 fragment layout
  #pragma unroll
  for (int kt = 0; kt < 2; kt++)
    #pragma unroll
    for (int e = 0; e < 16; e++) {
      int k = kt * 32 + ((e & 8) ? 16 : 0) + hh * 8 + ((e >> 1) & 3) * 2 + (e & 1);
      a[kt][e] = __builtin_bit_cast(__bf16, tile[r * D + k]);
    }

  int n = lane & 15;
  #pragma unroll
  for (int pj = 0; pj < 2; pj++) {
    int wofs = D + pj * D;
    unsigned short* outp = pj ? Vimg : Kimg;
    #pragma unroll
    for (int nt = 0; nt < 4; nt++) {
      float bias = ia_b[wofs + nt * 16 + n];
      v8f c;
      #pragma unroll
      for (int i = 0; i < 8; i++) c[i] = bias;
      #pragma unroll
      for (int kt = 0; kt < 2; kt++) {
        v16bf bf;   // B 32x16 fragment: k = (e&8 ? 16 : 0) + half*8 + (e&7)
        #pragma unroll
        for (int e = 0; e < 16; e++)
          bf[e] = f2bfh(ia_w[(wofs + nt * 16 + n) * D + kt * 32 + ((e & 8) ? 16 : 0) + hh * 8 + (e & 7)]);
        c = __builtin_amdgcn_wmma_f32_16x16x32_bf16(false, a[kt], false, bf,
                                                    (short)0, c, false, false);
      }
      #pragma unroll
      for (int rr = 0; rr < 8; rr++)
        outp[(row0 + rr + hh * 8) * D + nt * 16 + n] = f2bf(c[rr]);
    }
  }
}

// ============================================================================
// Kernel 3 (fused per-step): self-attn (flash, 2 waves/head, linear pos trick)
//   -> out-proj -> img q-proj -> img attention over SENC=1024 with
//   double-buffered GLOBAL_LOAD_ASYNC_TO_LDS staging of K -> epilogue
//   (out-proj, elu, rnn, skip, layernorm, token-embed, Hk/Hv cache update).
// One block (8 waves) per batch item; one launch per step.
// ============================================================================
__global__ void __launch_bounds__(256) k_step(
    const unsigned short* __restrict__ Kimg, const unsigned short* __restrict__ Vimg,
    const float* __restrict__ posk, const float* __restrict__ posv,
    const float* __restrict__ sa_w, const float* __restrict__ sa_b,
    const float* __restrict__ sa_wo, const float* __restrict__ sa_bo,
    const float* __restrict__ ia_w, const float* __restrict__ ia_b,
    const float* __restrict__ ia_wo, const float* __restrict__ ia_bo,
    const float* __restrict__ rnn_w, const float* __restrict__ rnn_b,
    const float* __restrict__ ln_g,  const float* __restrict__ ln_b,
    const float* __restrict__ tok_emb, const int* __restrict__ ids,
    float* __restrict__ H, float* __restrict__ Hk, float* __restrict__ Hv,
    float* __restrict__ hs, int t)
{
  __shared__ float hrow[D], q[D], head_out[D], sa_out[D];
  __shared__ float s_hm[8], s_hl[8], s_hacc[8][16];
  __shared__ float s_m[8], s_l[8], s_acc[8][D];
  __shared__ float attn[D], xe[D], y[D], hd_s[D], red[2];
  __shared__ __align__(16) unsigned short kbuf[2][8][16 * D];  // 32 KB K staging

  int b = blockIdx.x, tid = threadIdx.x;
  int w = tid >> 5, lane = tid & 31;

  // ---- Phase A: q projection of h_last (scale 1/sqrt(16) folded) ----
  if (tid < D) hrow[tid] = H[(size_t)b * SLOTS * D + (size_t)t * D + tid];
  __syncthreads();
  if (tid < D) {
    float s = sa_b[tid];
    for (int k = 0; k < D; k++) s += hrow[k] * sa_w[tid * D + k];
    q[tid] = s * 0.25f;
  }
  __syncthreads();

  // ---- Phase B: self-attention, 2 waves per head (hd=16), online softmax ----
  {
    int h = w >> 1, par = w & 1;
    const float* Hkb = Hk + (size_t)b * SLOTS * D + h * 16;
    const float* Hvb = Hv + (size_t)b * SLOTS * D + h * 16;
    float qh[16];
    #pragma unroll
    for (int d = 0; d < 16; d++) qh[d] = q[h * 16 + d];

    float m = -3e38f, l = 0.f, acc[16];
    #pragma unroll
    for (int d = 0; d < 16; d++) acc[d] = 0.f;

    for (int j = par * 32 + lane; j <= t; j += 64) {
      int pidx = 255 - t + j;   // clip(L-t-1+j,0,L-1): always in-range here
      const float* kp = Hkb + (size_t)j * D;
      const float* pk = posk + pidx * D + h * 16;
      float s = 0.f;
      #pragma unroll
      for (int d = 0; d < 16; d++) s += qh[d] * (kp[d] + pk[d]);
      float mn = fmaxf(m, s);
      float sc = __expf(m - mn);
      float p  = __expf(s - mn);
      l = l * sc + p;
      const float* vp = Hvb + (size_t)j * D;
      const float* pv = posv + pidx * D + h * 16;
      #pragma unroll
      for (int d = 0; d < 16; d++) acc[d] = acc[d] * sc + p * (vp[d] + pv[d]);
      m = mn;
    }
    #pragma unroll
    for (int off = 16; off; off >>= 1) {     // symmetric flash merge in-wave
      float m2 = __shfl_xor(m, off, 32);
      float l2 = __shfl_xor(l, off, 32);
      float mn = fmaxf(m, m2);
      float s1 = (l  > 0.f) ? __expf(m  - mn) : 0.f;
      float s2 = (l2 > 0.f) ? __expf(m2 - mn) : 0.f;
      #pragma unroll
      for (int d = 0; d < 16; d++) {
        float o = __shfl_xor(acc[d], off, 32);
        acc[d] = acc[d] * s1 + o * s2;
      }
      l = l * s1 + l2 * s2;
      m = mn;
    }
    if (lane == 0) {
      s_hm[w] = m; s_hl[w] = l;
      #pragma unroll
      for (int d = 0; d < 16; d++) s_hacc[w][d] = acc[d];
    }
  }
  __syncthreads();
  if (tid < D) {                              // merge the 2 wave partials per head
    int h = tid >> 4, d = tid & 15, w0 = 2 * h, w1 = 2 * h + 1;
    float m0 = s_hm[w0], m1 = s_hm[w1];
    float mt = fmaxf(m0, m1);
    float e0 = (s_hl[w0] > 0.f) ? __expf(m0 - mt) : 0.f;
    float e1 = (s_hl[w1] > 0.f) ? __expf(m1 - mt) : 0.f;
    float lt = s_hl[w0] * e0 + s_hl[w1] * e1;
    head_out[h * 16 + d] = (s_hacc[w0][d] * e0 + s_hacc[w1][d] * e1) / lt;
  }
  __syncthreads();

  // ---- Phase C: self-attn out-proj, then chained img q-proj (1/8 folded) ----
  if (tid < D) {
    float s = sa_bo[tid];
    for (int k = 0; k < D; k++) s += head_out[k] * sa_wo[tid * D + k];
    sa_out[tid] = s;
  }
  __syncthreads();
  if (tid < D) {
    float s = ia_b[tid];
    for (int k = 0; k < D; k++) s += sa_out[k] * ia_w[tid * D + k];
    q[tid] = s * 0.125f;                      // reuse q[] for the image query
  }
  __syncthreads();

  // ---- Phase E: image attention, async-LDS double-buffered K streaming ----
  {
    const unsigned short* Kb = Kimg + (size_t)b * SENC * D;
    const unsigned short* Vb = Vimg + (size_t)b * SENC * D;
    float m = -3e38f, l = 0.f, a0 = 0.f, a1 = 0.f;
    int jbase = w * 128;
    int rr = lane & 15, hh = lane >> 4;

    // issue chunk 0 (16 rows = 2 KB, 4x b128/lane, fully coalesced)
    #pragma unroll
    for (int i = 0; i < 4; i++)
      async_cp16(Kb + (size_t)jbase * D + (i * 32 + lane) * 8,
                 &kbuf[0][w][(i * 32 + lane) * 8]);

    int buf = 0;
    for (int c = 0; c < 8; c++) {
      if (c < 7) {
        int jn = jbase + (c + 1) * 16;
        __builtin_prefetch(Vb + (size_t)jn * D + lane * 32, 0, 0);
        #pragma unroll
        for (int i = 0; i < 4; i++)
          async_cp16(Kb + (size_t)jn * D + (i * 32 + lane) * 8,
                     &kbuf[buf ^ 1][w][(i * 32 + lane) * 8]);
        wait_async<4>();      // chunk c complete; chunk c+1 in flight
      } else {
        wait_async<0>();
      }
      // score for row rr of this chunk; lane halves split the 64-dim dot
      const unsigned short* kr = &kbuf[buf][w][rr * D + hh * 32];
      float s = 0.f;
      #pragma unroll
      for (int k8 = 0; k8 < 4; k8++) {
        uint4 u = *(const uint4*)(kr + k8 * 8);
        int k0 = hh * 32 + k8 * 8;
        s += q[k0 + 0] * lo16f(u.x) + q[k0 + 1] * hi16f(u.x)
           + q[k0 + 2] * lo16f(u.y) + q[k0 + 3] * hi16f(u.y)
           + q[k0 + 4] * lo16f(u.z) + q[k0 + 5] * hi16f(u.z)
           + q[k0 + 6] * lo16f(u.w) + q[k0 + 7] * hi16f(u.w);
      }
      s += __shfl_xor(s, 16, 32);             // both halves now hold the score
      float mblk = wred_max(s);
      float mn = fmaxf(m, mblk);
      float sc = __expf(m - mn);
      float p  = __expf(s - mn);
      l = l * sc + wred_sum(lane < 16 ? p : 0.f);
      a0 *= sc; a1 *= sc;
      int jrow0 = jbase + c * 16;
      #pragma unroll
      for (int jj = 0; jj < 16; jj++) {
        float pj = __shfl(p, jj, 32);
        unsigned uv = *(const unsigned*)(Vb + (size_t)(jrow0 + jj) * D + 2 * lane);
        a0 += pj * lo16f(uv);
        a1 += pj * hi16f(uv);
      }
      m = mn;
      buf ^= 1;
    }
    if (lane == 0) { s_m[w] = m; s_l[w] = l; }
    s_acc[w][2 * lane]     = a0;
    s_acc[w][2 * lane + 1] = a1;
  }
  __syncthreads();

  if (tid < D) {                              // merge the 8 wave partials
    float mt = s_m[0];
    #pragma unroll
    for (int ww = 1; ww < 8; ww++) mt = fmaxf(mt, s_m[ww]);
    float lt = 0.f, ot = 0.f;
    #pragma unroll
    for (int ww = 0; ww < 8; ww++) {
      float e = __expf(s_m[ww] - mt);
      lt += s_l[ww] * e;
      ot += s_acc[ww][tid] * e;
    }
    attn[tid] = ot / lt;
  }
  __syncthreads();
  if (tid < D) {                              // img out-proj + elu
    float s = ia_bo[tid];
    for (int k = 0; k < D; k++) s += attn[k] * ia_wo[tid * D + k];
    xe[tid] = s > 0.f ? s : __expf(s) - 1.f;
  }
  __syncthreads();
  if (tid < D) {                              // rnn + skip(h_last)
    float s = rnn_b[tid] + hrow[tid];
    for (int k = 0; k < D; k++) s += xe[k] * rnn_w[tid * D + k];
    y[tid] = s;
  }
  __syncthreads();
  if (tid < 32) {                             // layernorm stats
    float va = y[2 * tid], vb = y[2 * tid + 1];
    float s  = wred_sum(va + vb);
    float sq = wred_sum(va * va + vb * vb);
    if (tid == 0) {
      float mean = s / 64.f;
      red[0] = mean;
      red[1] = rsqrtf(sq / 64.f - mean * mean + 1e-5f);
    }
  }
  __syncthreads();
  if (tid < D) {
    float hn = (y[tid] - red[0]) * red[1] * ln_g[tid] + ln_b[tid];
    hs[((size_t)b * T + t) * D + tid] = hn;
    int id = ids[b * T + t];
    float hd = hn + tok_emb[id * D + tid];
    H[(size_t)b * SLOTS * D + (size_t)(t + 1) * D + tid] = hd;
    hd_s[tid] = hd;
  }
  __syncthreads();
  if (tid < 128) {                            // incremental K/V cache for new slot
    int n = tid & 63, isv = tid >> 6;
    float s = sa_b[D + isv * D + n];
    for (int k = 0; k < D; k++) s += hd_s[k] * sa_w[(D + isv * D + n) * D + k];
    (isv ? Hv : Hk)[(size_t)b * SLOTS * D + (size_t)(t + 1) * D + n] = s;
  }
}

// ============================================================================
// Kernel 4: final head (WMMA). logits = elu(hs@fc1^T+b1)@fc2^T+b2, elu fused.
// ============================================================================
__global__ void __launch_bounds__(32) k_final(
    const float* __restrict__ hs,
    const float* __restrict__ fc1_w, const float* __restrict__ fc1_b,
    const float* __restrict__ fc2_w, const float* __restrict__ fc2_b,
    float* __restrict__ out)
{
  __shared__ unsigned short xt[16 * D];
  int lane = threadIdx.x;
  int r = lane & 15, hh = lane >> 4, n = r;
  size_t row0 = (size_t)blockIdx.x * 16;

  v16bf a[2];
  #pragma unroll
  for (int kt = 0; kt < 2; kt++)
    #pragma unroll
    for (int e = 0; e < 16; e++) {
      int k = kt * 32 + ((e & 8) ? 16 : 0) + hh * 8 + ((e >> 1) & 3) * 2 + (e & 1);
      a[kt][e] = f2bfh(hs[(row0 + r) * D + k]);
    }

  #pragma unroll
  for (int nt = 0; nt < 4; nt++) {            // fc1: N=64
    float bias = fc1_b[nt * 16 + n];
    v8f c;
    #pragma unroll
    for (int i = 0; i < 8; i++) c[i] = bias;
    #pragma unroll
    for (int kt = 0; kt < 2; kt++) {
      v16bf bf;
      #pragma unroll
      for (int e = 0; e < 16; e++)
        bf[e] = f2bfh(fc1_w[(nt * 16 + n) * D + kt * 32 + ((e & 8) ? 16 : 0) + hh * 8 + (e & 7)]);
      c = __builtin_amdgcn_wmma_f32_16x16x32_bf16(false, a[kt], false, bf,
                                                  (short)0, c, false, false);
    }
    #pragma unroll
    for (int rr = 0; rr < 8; rr++) {
      float v = c[rr];
      v = v > 0.f ? v : __expf(v) - 1.f;      // fused elu
      xt[(rr + hh * 8) * D + nt * 16 + n] = f2bf(v);
    }
  }
  __syncthreads();

  v16bf a2[2];
  #pragma unroll
  for (int kt = 0; kt < 2; kt++)
    #pragma unroll
    for (int e = 0; e < 16; e++) {
      int k = kt * 32 + ((e & 8) ? 16 : 0) + hh * 8 + ((e >> 1) & 3) * 2 + (e & 1);
      a2[kt][e] = __builtin_bit_cast(__bf16, xt[r * D + k]);
    }

  #pragma unroll
  for (int nt = 0; nt < 8; nt++) {            // fc2: N=128
    float bias = fc2_b[nt * 16 + n];
    v8f c;
    #pragma unroll
    for (int i = 0; i < 8; i++) c[i] = bias;
    #pragma unroll
    for (int kt = 0; kt < 2; kt++) {
      v16bf bf;
      #pragma unroll
      for (int e = 0; e < 16; e++)
        bf[e] = f2bfh(fc2_w[(nt * 16 + n) * D + kt * 32 + ((e & 8) ? 16 : 0) + hh * 8 + (e & 7)]);
      c = __builtin_amdgcn_wmma_f32_16x16x32_bf16(false, a2[kt], false, bf,
                                                  (short)0, c, false, false);
    }
    #pragma unroll
    for (int rr = 0; rr < 8; rr++)
      out[(row0 + rr + hh * 8) * C + nt * 16 + n] = c[rr];
  }
}

// ============================================================================
// Host launcher. Workspace ~202 MB (bf16 K/V caches dominate).
// ============================================================================
extern "C" void kernel_launch(void* const* d_in, const int* in_sizes, int n_in,
                              void* d_out, int out_size, void* d_ws, size_t ws_size,
                              hipStream_t stream)
{
  const float* enc      = (const float*)d_in[0];
  const int*   ids      = (const int*)  d_in[1];
  const float* ln_enc_g = (const float*)d_in[2];
  const float* ln_enc_b = (const float*)d_in[3];
  const float* tok_emb  = (const float*)d_in[4];
  const float* pos      = (const float*)d_in[5];
  const float* sa_w     = (const float*)d_in[6];
  const float* sa_b     = (const float*)d_in[7];
  const float* sa_wo    = (const float*)d_in[8];
  const float* sa_bo    = (const float*)d_in[9];
  const float* ia_w     = (const float*)d_in[10];
  const float* ia_b     = (const float*)d_in[11];
  const float* ia_wo    = (const float*)d_in[12];
  const float* ia_bo    = (const float*)d_in[13];
  const float* h0       = (const float*)d_in[14];
  const float* fc1_w    = (const float*)d_in[15];
  const float* fc1_b    = (const float*)d_in[16];
  const float* fc2_w    = (const float*)d_in[17];
  const float* fc2_b    = (const float*)d_in[18];
  const float* rnn_w    = (const float*)d_in[19];
  const float* rnn_b    = (const float*)d_in[20];
  const float* ln_h_g   = (const float*)d_in[21];
  const float* ln_h_b   = (const float*)d_in[22];

  char* ws = (char*)d_ws;
  size_t o = 0;
  unsigned short* Kimg = (unsigned short*)(ws + o); o += (size_t)Bsz * SENC * D * 2;
  unsigned short* Vimg = (unsigned short*)(ws + o); o += (size_t)Bsz * SENC * D * 2;
  float* H    = (float*)(ws + o); o += (size_t)Bsz * SLOTS * D * 4;
  float* Hk   = (float*)(ws + o); o += (size_t)Bsz * SLOTS * D * 4;
  float* Hv   = (float*)(ws + o); o += (size_t)Bsz * SLOTS * D * 4;
  float* posk = (float*)(ws + o); o += (size_t)Lp * D * 4;
  float* posv = (float*)(ws + o); o += (size_t)Lp * D * 4;
  float* hsb  = (float*)(ws + o); o += (size_t)Bsz * T * D * 4;

  k_init<<<Lp + Bsz, 64, 0, stream>>>(pos, h0, sa_w, sa_b, posk, posv, H, Hk, Hv);
  k_encprep<<<(Bsz * SENC) / 16, 32, 0, stream>>>(enc, ln_enc_g, ln_enc_b, ia_w, ia_b, Kimg, Vimg);
  for (int t = 0; t < T; t++) {
    k_step<<<Bsz, 256, 0, stream>>>(Kimg, Vimg, posk, posv, sa_w, sa_b, sa_wo, sa_bo,
                                    ia_w, ia_b, ia_wo, ia_bo, rnn_w, rnn_b,
                                    ln_h_g, ln_h_b, tok_emb, ids, H, Hk, Hv, hsb, t);
  }
  k_final<<<(Bsz * T) / 16, 32, 0, stream>>>(hsb, fc1_w, fc1_b, fc2_w, fc2_b, (float*)d_out);
}
